// whole_network_48490180772311
// MI455X (gfx1250) — matrix operand order repaired and checked by reading, not compiled, not run on hardware
//
#include <hip/hip_runtime.h>

// ---------------- CDNA5 WMMA types ----------------
typedef __attribute__((ext_vector_type(16))) __bf16 v16bf;
typedef __attribute__((ext_vector_type(8)))  __bf16 v8bf;
typedef __attribute__((ext_vector_type(8)))  float  v8f;

#define WAVE 32

constexpr int NB = 128;          // batch
constexpr int NH = 8;            // heads
constexpr int NR = 130;          // relations
constexpr int NT = 256;          // features
constexpr int NQ = 32;           // per-head dim
constexpr int NRP = 160;         // NR padded to multiple of 32
constexpr int MPROJ = NB * NR;   // 16640
constexpr int NGCN = 8;
constexpr int LDSLDA = NT + 8;   // padded LDS row stride (bank-conflict-free, 16B aligned)

// ---------------- bf16 helpers ----------------
__device__ __forceinline__ unsigned short f2bf(float f) {
  unsigned int u = __builtin_bit_cast(unsigned int, f);
  u += 0x7FFFu + ((u >> 16) & 1u);
  return (unsigned short)(u >> 16);
}
__device__ __forceinline__ float bf2f(unsigned short u) {
  return __builtin_bit_cast(float, ((unsigned int)u) << 16);
}
__device__ __forceinline__ __bf16 bfraw(unsigned short u) {
  return __builtin_bit_cast(__bf16, u);
}

__device__ __forceinline__ v16bf frag_cat(v8bf a, v8bf b) {
  return __builtin_shufflevector(a, b, 0,1,2,3,4,5,6,7,8,9,10,11,12,13,14,15);
}

// A fragment: 16x32 tile at (m0,k0) of row-major [*,lda]; 2x16B loads per lane.
__device__ __forceinline__ v16bf load_a(const unsigned short* A, int lda,
                                        int m0, int k0, int lane) {
  int l = lane & 15, hi = lane >> 4;
  const unsigned short* p = A + (size_t)(m0 + l) * lda + (k0 + 8 * hi);
  v8bf x0 = *(const v8bf*)p;
  v8bf x1 = *(const v8bf*)(p + 16);
  return frag_cat(x0, x1);
}

// B fragment: 32x16 tile at (k0,n0); column n stored K-contiguously at Bp + n*sN.
__device__ __forceinline__ v16bf load_b(const unsigned short* Bp, size_t sN,
                                        int k0, int n0, int lane) {
  int l = lane & 15, hi = lane >> 4;
  const unsigned short* p = Bp + (size_t)(n0 + l) * sN + (k0 + 8 * hi);
  v8bf x0 = *(const v8bf*)p;
  v8bf x1 = *(const v8bf*)(p + 16);
  return frag_cat(x0, x1);
}

#define WMMA_BF16(a, b, c) \
  __builtin_amdgcn_wmma_f32_16x16x32_bf16(false, (a), false, (b), (short)0, (c), false, false)

// ---------------- utility kernels ----------------
__global__ void k_zero_u32(unsigned int* p, int n) {
  int i = blockIdx.x * blockDim.x + threadIdx.x;
  if (i < n) p[i] = 0u;
}

__global__ void k_f32_to_bf16(const float* __restrict__ s, unsigned short* __restrict__ d, int n) {
  int i = blockIdx.x * blockDim.x + threadIdx.x;
  if (i < n) d[i] = f2bf(s[i]);
}

// featsT[bh][t*NRP + r] = (r<NR) ? x[b,r,t] : 0
__global__ void k_featsT_init(const float* __restrict__ x, unsigned short* __restrict__ featsT) {
  int i = blockIdx.x * blockDim.x + threadIdx.x;
  int total = NB * NH * NT * NRP;
  if (i >= total) return;
  int r  = i % NRP;
  int t  = (i / NRP) % NT;
  int bh = i / (NRP * NT);
  int b  = bh >> 3;
  unsigned short v = 0;
  if (r < NR) v = f2bf(x[((size_t)(b * NR + r)) * NT + t]);
  featsT[i] = v;
}

// ---------------- projections: q,k (2M x 2N, 8 wmma per K-step) ----------------
__global__ __launch_bounds__(WAVE) void k_proj(const unsigned short* __restrict__ xb,
                                               const unsigned short* __restrict__ wqb,
                                               const unsigned short* __restrict__ wkb,
                                               const float* __restrict__ bq,
                                               const float* __restrict__ bk,
                                               unsigned short* __restrict__ qp,
                                               unsigned short* __restrict__ kp) {
  int lane = threadIdx.x;
  int m0 = blockIdx.x * 32;
  int n0 = blockIdx.y * 32;
  v8f cq[2][2] = {}, ck[2][2] = {};
  for (int k0 = 0; k0 < NT; k0 += 32) {
    v16bf a0  = load_a(xb, NT, m0,      k0, lane);
    v16bf a1  = load_a(xb, NT, m0 + 16, k0, lane);
    v16bf bq0 = load_b(wqb, (size_t)NT, k0, n0,      lane);
    v16bf bq1 = load_b(wqb, (size_t)NT, k0, n0 + 16, lane);
    v16bf bk0 = load_b(wkb, (size_t)NT, k0, n0,      lane);
    v16bf bk1 = load_b(wkb, (size_t)NT, k0, n0 + 16, lane);
    cq[0][0] = WMMA_BF16(a0, bq0, cq[0][0]);
    cq[0][1] = WMMA_BF16(a0, bq1, cq[0][1]);
    cq[1][0] = WMMA_BF16(a1, bq0, cq[1][0]);
    cq[1][1] = WMMA_BF16(a1, bq1, cq[1][1]);
    ck[0][0] = WMMA_BF16(a0, bk0, ck[0][0]);
    ck[0][1] = WMMA_BF16(a0, bk1, ck[0][1]);
    ck[1][0] = WMMA_BF16(a1, bk0, ck[1][0]);
    ck[1][1] = WMMA_BF16(a1, bk1, ck[1][1]);
  }
  int l = lane & 15, hi = lane >> 4;
#pragma unroll
  for (int nt = 0; nt < 2; ++nt) {
    int col = n0 + 16 * nt + l;
    float bqv = bq[col], bkv = bk[col];
#pragma unroll
    for (int mt = 0; mt < 2; ++mt) {
#pragma unroll
      for (int v = 0; v < 8; ++v) {
        unsigned row = m0 + 16 * mt + v + 8 * hi;   // compact row in [0, MPROJ)
        unsigned b = row / (unsigned)NR;
        unsigned r = row - b * (unsigned)NR;
        size_t base = ((size_t)b * NRP + r) * NT;
        qp[base + col] = f2bf(cq[mt][nt][v] + bqv);
        kp[base + col] = f2bf(ck[mt][nt][v] + bkv);
      }
    }
  }
}

// ---------------- rel = q @ k^T / sqrt(Q) per (b,h), 2x2 tiles ----------------
__global__ __launch_bounds__(WAVE) void k_rel(const unsigned short* __restrict__ qp,
                                              const unsigned short* __restrict__ kp,
                                              float* __restrict__ rel) {
  int lane = threadIdx.x;
  int m0 = blockIdx.x * 32, n0 = blockIdx.y * 32;
  int bh = blockIdx.z;
  int b = bh >> 3, h = bh & 7;
  const unsigned short* qbase = qp + (size_t)b * NRP * NT + h * NQ;
  const unsigned short* kbase = kp + (size_t)b * NRP * NT + h * NQ;
  v16bf a0 = load_a(qbase, NT, m0,      0, lane);
  v16bf a1 = load_a(qbase, NT, m0 + 16, 0, lane);
  v16bf b0 = load_b(kbase, (size_t)NT, 0, n0,      lane);
  v16bf b1 = load_b(kbase, (size_t)NT, 0, n0 + 16, lane);
  v8f c[2][2] = {};
  c[0][0] = WMMA_BF16(a0, b0, c[0][0]);
  c[0][1] = WMMA_BF16(a0, b1, c[0][1]);
  c[1][0] = WMMA_BF16(a1, b0, c[1][0]);
  c[1][1] = WMMA_BF16(a1, b1, c[1][1]);
  const float scale = 0.17677669529663687f;   // 1/sqrt(32)
  int l = lane & 15, hi = lane >> 4;
  float* rbase = rel + (size_t)bh * NR * NR;
#pragma unroll
  for (int nt = 0; nt < 2; ++nt) {
    int col = n0 + 16 * nt + l;
    if (col >= NR) continue;
#pragma unroll
    for (int mt = 0; mt < 2; ++mt) {
#pragma unroll
      for (int v = 0; v < 8; ++v) {
        int row = m0 + 16 * mt + v + 8 * hi;
        if (row < NR) rbase[(size_t)row * NR + col] = c[mt][nt][v] * scale;
      }
    }
  }
}

// ---------------- softmax -> relu(p-thr); gap accumulation ----------------
__global__ __launch_bounds__(WAVE) void k_softmax(float* __restrict__ rel,
                                                  float* __restrict__ gap,
                                                  const float* __restrict__ thrp) {
  int row = blockIdx.x;
  int bh = row / NR;
  float* rp = rel + (size_t)row * NR;
  int lane = threadIdx.x;
  float thr = thrp[0];
  bool tail = lane < (NR - 128);   // lanes 0,1 handle j=128,129
  float xs[5];
  float m = -3.0e38f;
#pragma unroll
  for (int it = 0; it < 4; ++it) { xs[it] = rp[lane + 32 * it]; m = fmaxf(m, xs[it]); }
  xs[4] = tail ? rp[lane + 128] : -3.0e38f;
  m = fmaxf(m, xs[4]);
  for (int o = 16; o; o >>= 1) m = fmaxf(m, __shfl_xor(m, o));
  float s = 0.f;
#pragma unroll
  for (int it = 0; it < 5; ++it) {
    float e = (it < 4 || tail) ? __expf(xs[it] - m) : 0.f;
    xs[it] = e;
    s += e;
  }
  for (int o = 16; o; o >>= 1) s += __shfl_xor(s, o);
  float inv = 1.f / s;
  float acc = 0.f;
#pragma unroll
  for (int it = 0; it < 4; ++it) {
    float v = fmaxf(xs[it] * inv - thr, 0.f);
    rp[lane + 32 * it] = v;
    acc += v;
  }
  if (tail) {
    float v = fmaxf(xs[4] * inv - thr, 0.f);
    rp[lane + 128] = v;
    acc += v;
  }
  for (int o = 16; o; o >>= 1) acc += __shfl_xor(acc, o);
  if (lane == 0) atomicAdd(gap + bh, acc);
}

// ---------------- SE gate ----------------
__global__ void k_se(const float* __restrict__ gap,
                     const float* __restrict__ W1, const float* __restrict__ b1,
                     const float* __restrict__ W2, const float* __restrict__ b2,
                     float* __restrict__ gate) {
  int b = blockIdx.x * blockDim.x + threadIdx.x;
  if (b >= NB) return;
  const float inv = 1.f / (float)(NR * NR);
  float g[NH], hd[NH / 2];
#pragma unroll
  for (int h = 0; h < NH; ++h) g[h] = gap[b * NH + h] * inv;
#pragma unroll
  for (int o = 0; o < NH / 2; ++o) {
    float s = b1[o];
#pragma unroll
    for (int h = 0; h < NH; ++h) s += W1[o * NH + h] * g[h];
    hd[o] = fmaxf(s, 0.f);
  }
#pragma unroll
  for (int h = 0; h < NH; ++h) {
    float s = b2[h];
#pragma unroll
    for (int o = 0; o < NH / 2; ++o) s += W2[h * (NH / 2) + o] * hd[o];
    float sg = 1.f / (1.f + __expf(-s));
    gate[b * NH + h] = floorf(sg);
  }
}

// ---------------- gate + rank mask -> padded bf16 adjacency ----------------
__global__ __launch_bounds__(WAVE) void k_mask(const float* __restrict__ rel,
                                               const float* __restrict__ gate,
                                               unsigned short* __restrict__ Apad) {
  int row = blockIdx.x;
  int bh = row / NR, r = row % NR;
  const float* rp = rel + (size_t)row * NR;
  unsigned short* ap = Apad + (size_t)bh * NRP * NRP + (size_t)r * NRP;
  float g = gate[bh];
  int lane = threadIdx.x;
  bool tail = lane < (NR - 128);
  const int refc[6] = {0, 4, 5, 6, 7, 8};
  float refv[6];
#pragma unroll
  for (int t = 0; t < 6; ++t) refv[t] = rp[refc[t]] * g;
  int cnt[6] = {0, 0, 0, 0, 0, 0};
  float xs[5];
#pragma unroll
  for (int it = 0; it < 4; ++it) xs[it] = rp[lane + 32 * it] * g;
  xs[4] = tail ? rp[lane + 128] * g : 0.f;
#pragma unroll
  for (int it = 0; it < 5; ++it) {
    int j = lane + 32 * it;
    if (it < 4 || tail) {
      float x = xs[it];
#pragma unroll
      for (int t = 0; t < 6; ++t)
        cnt[t] += (int)(x > refv[t]) + (int)((x == refv[t]) && (j < refc[t]));
    }
  }
#pragma unroll
  for (int t = 0; t < 6; ++t) {
    int c = cnt[t];
    for (int o = 16; o; o >>= 1) c += __shfl_xor(c, o);
    cnt[t] = c;   // stable descending-rank of reference column t
  }
#pragma unroll
  for (int it = 0; it < 5; ++it) {
    int j = lane + 32 * it;       // j in [0,160)
    bool keep = (j == r);
#pragma unroll
    for (int t = 0; t < 6; ++t) keep = keep || (j == cnt[t]);
    float v = (j < NR && (it < 4 || tail) && keep) ? xs[it] : 0.f;
    ap[j] = f2bf(v);
  }
}

// ---------------- fused GCN layer (32-row block, 2x2 tiles per wave) ----------------
// Stage 1: aggL[32,256] = A[m0:m0+32,:] @ feats (LDS, padded stride).
// Stage 2: dst = relu(aggL @ W^T + b) + src.  featsT double-buffered across layers.
__global__ __launch_bounds__(256) void k_gcn_layer(const unsigned short* __restrict__ Apad,
                                                   const unsigned short* __restrict__ srcT,
                                                   unsigned short* __restrict__ dstT,
                                                   const unsigned short* __restrict__ Wb,
                                                   const float* __restrict__ bias) {
  __shared__ unsigned short aggL[32 * LDSLDA];   // ~16.5KB, conflict-free stride
  int tid = threadIdx.x;
  int lane = tid & 31;
  int w = tid >> 5;                          // wave 0..7
  int m0 = blockIdx.x * 32;
  int bh = blockIdx.y;
  const unsigned short* Ab = Apad + (size_t)bh * NRP * NRP;
  const unsigned short* Fs = srcT + (size_t)bh * NT * NRP;
  unsigned short* Fd = dstT + (size_t)bh * NT * NRP;
  int l = lane & 15, hi = lane >> 4;
  int n0 = w * 16, n1 = n0 + 128;

  // ---- stage 1: agg 32-row block into LDS ----
  {
    v8f c[2][2] = {};
    for (int k0 = 0; k0 < NRP; k0 += 32) {
      v16bf a0 = load_a(Ab, NRP, m0,      k0, lane);
      v16bf a1 = load_a(Ab, NRP, m0 + 16, k0, lane);
      v16bf b0 = load_b(Fs, (size_t)NRP, k0, n0, lane);
      v16bf b1 = load_b(Fs, (size_t)NRP, k0, n1, lane);
      c[0][0] = WMMA_BF16(a0, b0, c[0][0]);
      c[0][1] = WMMA_BF16(a0, b1, c[0][1]);
      c[1][0] = WMMA_BF16(a1, b0, c[1][0]);
      c[1][1] = WMMA_BF16(a1, b1, c[1][1]);
    }
#pragma unroll
    for (int mt = 0; mt < 2; ++mt) {
#pragma unroll
      for (int v = 0; v < 8; ++v) {
        int row = 16 * mt + v + 8 * hi;
        aggL[row * LDSLDA + n0 + l] = f2bf(c[mt][0][v]);
        aggL[row * LDSLDA + n1 + l] = f2bf(c[mt][1][v]);
      }
    }
  }
  __syncthreads();

  // ---- stage 2: contract agg (LDS) with W^T; relu + residual; store dstT ----
  {
    v8f c[2][2] = {};
    for (int k0 = 0; k0 < NT; k0 += 32) {
      const unsigned short* pa0 = &aggL[l * LDSLDA + (k0 + 8 * hi)];
      const unsigned short* pa1 = &aggL[(16 + l) * LDSLDA + (k0 + 8 * hi)];
      v16bf a0 = frag_cat(*(const v8bf*)pa0, *(const v8bf*)(pa0 + 16));
      v16bf a1 = frag_cat(*(const v8bf*)pa1, *(const v8bf*)(pa1 + 16));
      v16bf b0 = load_b(Wb, (size_t)NT, k0, n0, lane);
      v16bf b1 = load_b(Wb, (size_t)NT, k0, n1, lane);
      c[0][0] = WMMA_BF16(a0, b0, c[0][0]);
      c[0][1] = WMMA_BF16(a0, b1, c[0][1]);
      c[1][0] = WMMA_BF16(a1, b0, c[1][0]);
      c[1][1] = WMMA_BF16(a1, b1, c[1][1]);
    }
#pragma unroll
    for (int nt = 0; nt < 2; ++nt) {
      int col = (nt == 0 ? n0 : n1) + l;
      float bv = bias[col];
#pragma unroll
      for (int mt = 0; mt < 2; ++mt) {
        const unsigned short* sp = Fs + (size_t)col * NRP + m0 + 16 * mt + 8 * hi;
        unsigned short* dp = Fd + (size_t)col * NRP + m0 + 16 * mt + 8 * hi;
        v8bf oldv = *(const v8bf*)sp;
        v8bf outv;
#pragma unroll
        for (int v = 0; v < 8; ++v) {
          float o = fmaxf(c[mt][nt][v] + bv, 0.f) +
                    bf2f(__builtin_bit_cast(unsigned short, oldv[v]));
          outv[v] = bfraw(f2bf(o));
        }
        *(v8bf*)dp = outv;
      }
    }
  }
}

// ---------------- fusion over heads + mean-pool over T ----------------
__global__ __launch_bounds__(NRP) void k_fuse_pool(const unsigned short* __restrict__ featsT,
                                                   const float* __restrict__ fw,
                                                   const float* __restrict__ fbias,
                                                   float* __restrict__ pooled) {
  int b = blockIdx.x;
  int r = threadIdx.x;
  if (r >= NR) return;
  float acc = 0.f;
  for (int h = 0; h < NH; ++h) {
    const unsigned short* fp = featsT + (size_t)(b * NH + h) * NT * NRP + r;
    float wv = fw[h];
#pragma unroll 8
    for (int t = 0; t < NT; ++t) acc += wv * bf2f(fp[(size_t)t * NRP]);
  }
  pooled[b * NR + r] = acc * (1.f / (float)NT) + fbias[0];
}

// ---------------- MLP head: 130 -> 64 -> 16 -> 5 ----------------
__global__ __launch_bounds__(64) void k_mlp(const float* __restrict__ pooled,
                                            const float* __restrict__ W1, const float* __restrict__ b1,
                                            const float* __restrict__ W2, const float* __restrict__ b2,
                                            const float* __restrict__ W3, const float* __restrict__ b3,
                                            float* __restrict__ out) {
  __shared__ float p[NR];
  __shared__ float h1[64];
  __shared__ float h2[16];
  int b = blockIdx.x, tid = threadIdx.x;
  for (int j = tid; j < NR; j += 64) p[j] = pooled[b * NR + j];
  __syncthreads();
  {
    float s = b1[tid];
    for (int j = 0; j < NR; ++j) s += W1[tid * NR + j] * p[j];
    h1[tid] = fmaxf(s, 0.f);
  }
  __syncthreads();
  if (tid < 16) {
    float s = b2[tid];
    for (int j = 0; j < 64; ++j) s += W2[tid * 64 + j] * h1[j];
    h2[tid] = fmaxf(s, 0.f);
  }
  __syncthreads();
  if (tid < 5) {
    float s = b3[tid];
    for (int j = 0; j < 16; ++j) s += W3[tid * 16 + j] * h2[j];
    out[b * 5 + tid] = fmaxf(s, 0.f);
  }
}

// ---------------- host launcher ----------------
extern "C" void kernel_launch(void* const* d_in, const int* in_sizes, int n_in,
                              void* d_out, int out_size, void* d_ws, size_t ws_size,
                              hipStream_t stream) {
  (void)in_sizes; (void)n_in; (void)out_size; (void)ws_size;
  const float* x    = (const float*)d_in[0];
  const float* Wq   = (const float*)d_in[1];
  const float* bq   = (const float*)d_in[2];
  const float* Wk   = (const float*)d_in[3];
  const float* bk   = (const float*)d_in[4];
  const float* thr  = (const float*)d_in[5];
  const float* seW1 = (const float*)d_in[6];
  const float* seb1 = (const float*)d_in[7];
  const float* seW2 = (const float*)d_in[8];
  const float* seb2 = (const float*)d_in[9];
  const float* gW   = (const float*)d_in[10];
  const float* gb   = (const float*)d_in[11];
  const float* fw   = (const float*)d_in[12];
  const float* fb   = (const float*)d_in[13];
  const float* mW1  = (const float*)d_in[14];
  const float* mb1  = (const float*)d_in[15];
  const float* mW2  = (const float*)d_in[16];
  const float* mb2  = (const float*)d_in[17];
  const float* mW3  = (const float*)d_in[18];
  const float* mb3  = (const float*)d_in[19];
  float* out = (float*)d_out;

  char* wsb = (char*)d_ws;
  size_t off = 0;
  auto carve = [&](size_t bytes) -> char* {
    char* p = wsb + off;
    off += (bytes + 255) & ~(size_t)255;
    return p;
  };
  unsigned short* xb    = (unsigned short*)carve((size_t)NB * NR * NT * 2);
  unsigned short* wqb   = (unsigned short*)carve((size_t)NT * NT * 2);
  unsigned short* wkb   = (unsigned short*)carve((size_t)NT * NT * 2);
  unsigned short* gwb   = (unsigned short*)carve((size_t)NGCN * NT * NT * 2);
  unsigned short* qp    = (unsigned short*)carve((size_t)NB * NRP * NT * 2);
  unsigned short* kp    = (unsigned short*)carve((size_t)NB * NRP * NT * 2);
  float*          rel   = (float*)carve((size_t)NB * NH * NR * NR * 4);
  float*          gap   = (float*)carve((size_t)NB * NH * 4);
  float*          gate  = (float*)carve((size_t)NB * NH * 4);
  unsigned short* Apad  = (unsigned short*)carve((size_t)NB * NH * NRP * NRP * 2);
  unsigned short* fTa   = (unsigned short*)carve((size_t)NB * NH * NT * NRP * 2);
  unsigned short* fTb   = (unsigned short*)carve((size_t)NB * NH * NT * NRP * 2);
  float*          pooled= (float*)carve((size_t)NB * NR * 4);

  // zero padded buffers + gap accumulator
  {
    int nw = (int)((size_t)NB * NRP * NT * 2 / 4);
    k_zero_u32<<<(nw + 255) / 256, 256, 0, stream>>>((unsigned int*)qp, nw);
    k_zero_u32<<<(nw + 255) / 256, 256, 0, stream>>>((unsigned int*)kp, nw);
  }
  {
    int nw = (int)((size_t)NB * NH * NRP * NRP * 2 / 4);
    k_zero_u32<<<(nw + 255) / 256, 256, 0, stream>>>((unsigned int*)Apad, nw);
  }
  k_zero_u32<<<(NB * NH + 255) / 256, 256, 0, stream>>>((unsigned int*)gap, NB * NH);

  // bf16 conversions
  {
    int n = NB * NR * NT;
    k_f32_to_bf16<<<(n + 255) / 256, 256, 0, stream>>>(x, xb, n);
  }
  k_f32_to_bf16<<<(NT * NT + 255) / 256, 256, 0, stream>>>(Wq, wqb, NT * NT);
  k_f32_to_bf16<<<(NT * NT + 255) / 256, 256, 0, stream>>>(Wk, wkb, NT * NT);
  k_f32_to_bf16<<<(NGCN * NT * NT + 255) / 256, 256, 0, stream>>>(gW, gwb, NGCN * NT * NT);

  // projections + relation matrix (WMMA)
  k_proj<<<dim3(MPROJ / 32, NT / 32), WAVE, 0, stream>>>(xb, wqb, wkb, bq, bk, qp, kp);
  {
    int mt = (NR + 31) / 32;   // 5 blocks of 32 cover 130 (reads stay in padded 160)
    k_rel<<<dim3(mt, mt, NB * NH), WAVE, 0, stream>>>(qp, kp, rel);
  }

  // softmax + threshold + gap; SE gate; rank mask -> adjacency
  k_softmax<<<NB * NH * NR, WAVE, 0, stream>>>(rel, gap, thr);
  k_se<<<(NB + 127) / 128, 128, 0, stream>>>(gap, seW1, seb1, seW2, seb2, gate);
  k_mask<<<NB * NH * NR, WAVE, 0, stream>>>(rel, gate, Apad);

  // feats init (transposed, padded, broadcast over heads)
  {
    int n = NB * NH * NT * NRP;
    k_featsT_init<<<(n + 255) / 256, 256, 0, stream>>>(x, fTa);
  }

  // 8 fused GCN layers, ping-pong featsT
  unsigned short* src = fTa;
  unsigned short* dst = fTb;
  for (int layer = 0; layer < NGCN; ++layer) {
    k_gcn_layer<<<dim3(NRP / 32, NB * NH), 256, 0, stream>>>(
        Apad, src, dst, gwb + (size_t)layer * NT * NT, gb + (size_t)layer * NT);
    unsigned short* tmp = src; src = dst; dst = tmp;
  }

  // fusion + pool, then MLP head  (after 8 swaps, result is in `src`)
  k_fuse_pool<<<NB, NRP, 0, stream>>>(src, fw, fb, pooled);
  k_mlp<<<NB, 64, 0, stream>>>(pooled, mW1, mb1, mW2, mb2, mW3, mb3, out);
}
